// SelfAttentionLayerFull_18657337933943
// MI455X (gfx1250) — matrix operand order repaired and checked by reading, not compiled, not run on hardware
//
#include <hip/hip_runtime.h>
#include <hip/hip_bf16.h>
#include <math.h>

typedef __attribute__((ext_vector_type(16))) _Float16 v16h;
typedef __attribute__((ext_vector_type(8)))  _Float16 v8h;
typedef __attribute__((ext_vector_type(8)))  float    v8f;

#define DEVINL static __device__ __forceinline__

constexpr int BATCH = 4;
constexpr int NTOK  = 4096;   // 8*8*8*8
constexpr int DIM   = 512;    // IN_DIM == CMP_DIM == OUT_DIM
constexpr int BK    = 128;    // keys per flash tile

// ---- WMMA fragment loaders (ISA 7.12.2 layouts, wave32) -------------------

// A fragment (16x32 f16, M x K): lane<16 -> row=lane, K {0..7,16..23};
// lane>=16 -> row=lane-16, K {8..15,24..31}. Two 16B loads.
DEVINL v16h loadA16(const _Float16* base, int ld) {
  const int lane = threadIdx.x & 31;
  const _Float16* p = base + (lane & 15) * ld + (lane >> 4) * 8;
  union { v16h v; v8h h[2]; } u;
  u.h[0] = *(const v8h*)(p);
  u.h[1] = *(const v8h*)(p + 16);
  return u.v;
}

// B fragment (32x16 f16, K x N) loaded from an N-major [n][k] matrix:
// lane<16 -> col=lane, K {0..7,16..23}; lane>=16 -> col=lane-16, K {8..15,24..31}.
DEVINL v16h loadB16(const _Float16* base, int ld) {
  const int lane = threadIdx.x & 31;
  const _Float16* p = base + (lane & 15) * ld + (lane >> 4) * 8;
  union { v16h v; v8h h[2]; } u;
  u.h[0] = *(const v8h*)(p);
  u.h[1] = *(const v8h*)(p + 16);
  return u.v;
}

// A fragment sourced from fp32 memory, converted to f16 in registers.
DEVINL v16h loadA32cvt(const float* base, int ld) {
  const int lane = threadIdx.x & 31;
  const float* p = base + (lane & 15) * ld + (lane >> 4) * 8;
  float4 a0 = *(const float4*)(p);
  float4 a1 = *(const float4*)(p + 4);
  float4 b0 = *(const float4*)(p + 16);
  float4 b1 = *(const float4*)(p + 20);
  v16h r;
  r[0]=(_Float16)a0.x;  r[1]=(_Float16)a0.y;  r[2]=(_Float16)a0.z;  r[3]=(_Float16)a0.w;
  r[4]=(_Float16)a1.x;  r[5]=(_Float16)a1.y;  r[6]=(_Float16)a1.z;  r[7]=(_Float16)a1.w;
  r[8]=(_Float16)b0.x;  r[9]=(_Float16)b0.y;  r[10]=(_Float16)b0.z; r[11]=(_Float16)b0.w;
  r[12]=(_Float16)b1.x; r[13]=(_Float16)b1.y; r[14]=(_Float16)b1.z; r[15]=(_Float16)b1.w;
  return r;
}

DEVINL v8f wmma16(v16h a, v16h b, v8f c) {
  return __builtin_amdgcn_wmma_f32_16x16x32_f16(false, a, false, b, (short)0, c,
                                                false, false);
}

// ---- Stage 1: W[k][n] fp32 -> Wt[n][k] f16 --------------------------------
__global__ void wt_transpose_kernel(const float* __restrict__ W,
                                    _Float16* __restrict__ Wt) {
  int idx = blockIdx.x * blockDim.x + threadIdx.x;  // 512*512 threads
  int k = idx >> 9;
  int n = idx & 511;
  Wt[(size_t)n * DIM + k] = (_Float16)W[idx];       // coalesced read
}

// ---- Stage 2: projection GEMM  Out = cast16((X @ W + b) * scale) ----------
// 8 waves, tile 128(M) x 64(N); wave w owns rows [w*16, w*16+16).
// transposeOut: 0 -> Out[m][d] (Q, K); 1 -> Out[batch][d][mq] (V transposed).
__global__ __launch_bounds__(256) void proj_kernel(
    const float* __restrict__ X, const _Float16* __restrict__ Wt,
    const float* __restrict__ bias, _Float16* __restrict__ Out,
    float scale, int transposeOut) {
  const int wave = threadIdx.x >> 5;
  const int lane = threadIdx.x & 31;
  const int half = lane >> 4;
  const int col  = lane & 15;
  const int m0 = blockIdx.x * 128 + wave * 16;
  const int n0 = blockIdx.y * 64;

  const float* Xr = X + (size_t)m0 * DIM;
  v8f acc[4] = {};
#pragma unroll
  for (int c = 0; c < 16; ++c) {               // K = 512 in steps of 32
    v16h af = loadA32cvt(Xr + c * 32, DIM);
#pragma unroll
    for (int n = 0; n < 4; ++n) {
      v16h bf = loadB16(Wt + (size_t)(n0 + n * 16) * DIM + c * 32, DIM);
      acc[n] = wmma16(af, bf, acc[n]);
    }
  }
#pragma unroll
  for (int n = 0; n < 4; ++n) {
    const int d = n0 + n * 16 + col;
    const float bv = bias[d];
#pragma unroll
    for (int r = 0; r < 8; ++r) {
      const int m = m0 + r + 8 * half;
      const float v = (acc[n][r] + bv) * scale;
      if (transposeOut) {
        const int bb = m >> 12, mq = m & (NTOK - 1);
        Out[((size_t)bb * DIM + d) * NTOK + mq] = (_Float16)v;
      } else {
        Out[(size_t)m * DIM + d] = (_Float16)v;
      }
    }
  }
}

// ---- Stage 3: flash attention ---------------------------------------------
// One WG (8 waves) per 16 query rows of one batch. Online softmax over 4096
// keys in tiles of 128. Wave w computes the 16x16 score subtile for keys
// [k0+16w, k0+16w+16) and accumulates output dims [64w, 64w+64).
__global__ __launch_bounds__(256) void flash_kernel(
    const _Float16* __restrict__ Qh,   // [b][n][c] f16, pre-scaled by 1/sqrt(c)
    const _Float16* __restrict__ Kh,   // [b][n][c] f16
    const _Float16* __restrict__ Vt,   // [b][d][n] f16 (transposed)
    float* __restrict__ out) {         // [b][n][d] f32
  __shared__ float    Sld[16 * BK];
  __shared__ _Float16 Pld[16 * BK];
  __shared__ float    red[16 * 16];
  __shared__ float    m_s[16], l_s[16], a_s[16];

  const int tid  = threadIdx.x;
  const int wave = tid >> 5;
  const int lane = tid & 31;
  const int half = lane >> 4;
  const int col  = lane & 15;
  const int b  = blockIdx.y;
  const int q0 = blockIdx.x * 16;

  const _Float16* Qb = Qh + ((size_t)b * NTOK + q0) * DIM;
  const _Float16* Kb = Kh + (size_t)b * NTOK * DIM;
  const _Float16* Vb = Vt + (size_t)b * DIM * NTOK;

  // keep the 16x512 Q strip resident in VGPRs (16 A-fragments)
  v16h qf[16];
#pragma unroll
  for (int c = 0; c < 16; ++c) qf[c] = loadA16(Qb + c * 32, DIM);

  v8f acc[4] = {};
  if (tid < 16) { m_s[tid] = -1e30f; l_s[tid] = 0.f; }
  __syncthreads();

  const int row = tid >> 4;   // 0..15
  const int seg = tid & 15;   // 0..15 -> 8 score columns each

  for (int k0 = 0; k0 < NTOK; k0 += BK) {
    // S subtile = Q (16x512) x K^T (keys 16w..16w+15); B-frag = K rows
    v8f s = {};
    const _Float16* Kt = Kb + (size_t)(k0 + wave * 16) * DIM;
#pragma unroll
    for (int c = 0; c < 16; ++c) {
      v16h kf = loadB16(Kt + c * 32, DIM);
      s = wmma16(qf[c], kf, s);
    }
#pragma unroll
    for (int r = 0; r < 8; ++r)
      Sld[(r + 8 * half) * BK + wave * 16 + col] = s[r];
    __syncthreads();

    // tile row-max: 16 threads/row, 8 cols each
    float pm = -1e30f;
#pragma unroll
    for (int j = 0; j < 8; ++j) pm = fmaxf(pm, Sld[row * BK + seg * 8 + j]);
    red[row * 16 + seg] = pm;
    __syncthreads();

    if (tid < 16) {
      float mt = -1e30f;
      for (int j = 0; j < 16; ++j) mt = fmaxf(mt, red[tid * 16 + j]);
      const float mo = m_s[tid];
      const float mn = fmaxf(mo, mt);
      a_s[tid] = __expf(mo - mn);
      m_s[tid] = mn;
    }
    __syncthreads();

    // rescale O accumulators by alpha
    {
      float al[8];
#pragma unroll
      for (int r = 0; r < 8; ++r) al[r] = a_s[r + 8 * half];
#pragma unroll
      for (int n = 0; n < 4; ++n)
#pragma unroll
        for (int r = 0; r < 8; ++r) acc[n][r] *= al[r];
    }

    // P = exp(S - m_new) -> f16 LDS; partial row sums
    {
      const float mn = m_s[row];
      float ps = 0.f;
#pragma unroll
      for (int j = 0; j < 8; ++j) {
        const int i = row * BK + seg * 8 + j;
        const float e = __expf(Sld[i] - mn);
        Pld[i] = (_Float16)e;
        ps += e;
      }
      red[row * 16 + seg] = ps;
    }
    __syncthreads();

    if (tid < 16) {
      float sm = 0.f;
      for (int j = 0; j < 16; ++j) sm += red[tid * 16 + j];
      l_s[tid] = l_s[tid] * a_s[tid] + sm;
    }

    // O += P (16x128, from LDS) @ V (keys x d); B-frag from Vt rows
#pragma unroll
    for (int cc = 0; cc < 4; ++cc) {
      v16h pf = loadA16((const _Float16*)Pld + cc * 32, BK);
#pragma unroll
      for (int n = 0; n < 4; ++n) {
        v16h vf = loadB16(Vb + (size_t)(wave * 64 + n * 16) * NTOK + k0 + cc * 32,
                          NTOK);
        acc[n] = wmma16(pf, vf, acc[n]);
      }
    }
    __syncthreads();
  }

  float linv[8];
#pragma unroll
  for (int r = 0; r < 8; ++r) linv[r] = 1.f / l_s[r + 8 * half];

  float* ob = out + ((size_t)b * NTOK + q0) * DIM + wave * 64;
#pragma unroll
  for (int n = 0; n < 4; ++n)
#pragma unroll
    for (int r = 0; r < 8; ++r)
      ob[(size_t)(r + 8 * half) * DIM + n * 16 + col] = acc[n][r] * linv[r];
}

// ---- host -----------------------------------------------------------------
extern "C" void kernel_launch(void* const* d_in, const int* in_sizes, int n_in,
                              void* d_out, int out_size, void* d_ws, size_t ws_size,
                              hipStream_t stream) {
  const float* query_X = (const float*)d_in[0];
  const float* key_X   = (const float*)d_in[1];
  const float* value_X = (const float*)d_in[2];
  const float* W_Q = (const float*)d_in[3];
  const float* b_Q = (const float*)d_in[4];
  const float* W_K = (const float*)d_in[5];
  const float* b_K = (const float*)d_in[6];
  const float* W_V = (const float*)d_in[7];
  const float* b_V = (const float*)d_in[8];
  float* out = (float*)d_out;

  // workspace layout (f16): Q, K, Vt, WtQ, WtK, WtV  (~50 MB total)
  const size_t QKV = (size_t)BATCH * NTOK * DIM;
  _Float16* Qh  = (_Float16*)d_ws;
  _Float16* Kh  = Qh + QKV;
  _Float16* Vt  = Kh + QKV;
  _Float16* WtQ = Vt + QKV;
  _Float16* WtK = WtQ + (size_t)DIM * DIM;
  _Float16* WtV = WtK + (size_t)DIM * DIM;

  wt_transpose_kernel<<<(DIM * DIM) / 256, 256, 0, stream>>>(W_Q, WtQ);
  wt_transpose_kernel<<<(DIM * DIM) / 256, 256, 0, stream>>>(W_K, WtK);
  wt_transpose_kernel<<<(DIM * DIM) / 256, 256, 0, stream>>>(W_V, WtV);

  dim3 pg((BATCH * NTOK) / 128, DIM / 64);
  const float rs = 0.044194173824159216f;  // 1/sqrt(512), folded into Q
  proj_kernel<<<pg, 256, 0, stream>>>(query_X, WtQ, b_Q, Qh, rs,   0);
  proj_kernel<<<pg, 256, 0, stream>>>(key_X,   WtK, b_K, Kh, 1.0f, 0);
  proj_kernel<<<pg, 256, 0, stream>>>(value_X, WtV, b_V, Vt, 1.0f, 1);

  dim3 fg(NTOK / 16, BATCH);
  flash_kernel<<<fg, 256, 0, stream>>>(Qh, Kh, Vt, out);
}
